// LSTM_64158221467796
// MI455X (gfx1250) — compile-verified, matrix operand order checked by reading, not run on hardware
//
#include <hip/hip_runtime.h>
#include <hip/hip_bf16.h>
#include <math.h>
#include <stdint.h>

// ---------------- problem constants ----------------
#define BB   16
#define TT   128
#define HH   512
#define VV   50257
#define VP   50272            // V padded to multiple of 16
#define NT_V (VP/16)          // 3142 n-tiles in logits GEMM
#define STRIP 32              // n-tiles per wave strip
#define NSTRIPS ((NT_V + STRIP - 1)/STRIP)   // 99
#define MTILES ((BB*TT)/16)   // 128 row tiles
#define ROWS (BB*TT)          // 2048

typedef __bf16 bf16;
typedef __attribute__((ext_vector_type(16))) __bf16 v16bf;
typedef __attribute__((ext_vector_type(8)))  float  v8f;
typedef __attribute__((ext_vector_type(4)))  unsigned int v4u;

// global (address_space(1)) pointer to a 16-byte vector: forces global_load_b128
// (not flat_load_b128, which would also tie up DScnt and the LDS port).
typedef const __attribute__((address_space(1))) v4u* gv4u_p;

// ---------------- fragment helpers ----------------
// 16-bit A/B fragment layout (CDNA5 ISA 7.12.2):
//  lanes 0-15 : row/col = lane,     K = kb+{0..7}  and kb+{16..23}
//  lanes 16-31: row/col = lane-16,  K = kb+{8..15} and kb+{24..31}
// With row-major (K-contiguous) storage this is two 16-byte loads per lane.
__device__ inline v16bf load_frag(const bf16* p /* row base + kb */, int half) {
  union { v4u u[2]; v16bf v; } r;
  r.u[0] = *(gv4u_p)(uintptr_t)(p + half * 8);
  r.u[1] = *(gv4u_p)(uintptr_t)(p + 16 + half * 8);
  return r.v;
}

__device__ inline v8f wmma_bf16(v16bf a, v16bf b, v8f c) {
  // D(f32 16x16) = A(16x32 bf16) x B(32x16 bf16) + C
  return __builtin_amdgcn_wmma_f32_16x16x32_bf16(
      /*neg_a=*/false, a, /*neg_b=*/false, b,
      /*c_mod=*/(short)0, c, /*reuse_a=*/false, /*reuse_b=*/false);
}

__device__ inline float sigm(float x) { return 1.0f / (1.0f + __expf(-x)); }

// monotone order-preserving float->uint key (for atomic float max via atomicMax u32)
__device__ inline unsigned encKey(float f) {
  unsigned b = __float_as_uint(f);
  return (b & 0x80000000u) ? ~b : (b | 0x80000000u);
}

// ---------------- prep kernels ----------------
__global__ void k_init_stats(unsigned* rowmaxU, float* rowsum, float* out) {
  int i = blockIdx.x * 256 + threadIdx.x;
  if (i < ROWS) { rowmaxU[i] = encKey(-3.0e38f); rowsum[i] = 0.0f; }
  if (i == 0) out[0] = 0.0f;
}

// W [2H,4H] f32 row-major -> Wt [4H][2H] bf16 (N-major, K contiguous)
__global__ void k_transpose_w(const float* __restrict__ W, bf16* __restrict__ Wt) {
  int idx = blockIdx.x * 256 + threadIdx.x;      // 2048*1024 total
  int g = idx >> 10, k = idx & 1023;
  Wt[(size_t)g * 1024 + k] = (bf16)W[(size_t)k * 2048 + g];
}

// softmax_w [H,V] f32 -> Wsmt [VP][H] bf16 (zero-padded rows for n >= V)
__global__ void k_transpose_sm(const float* __restrict__ SW, bf16* __restrict__ Wt) {
  int idx = blockIdx.x * 256 + threadIdx.x;      // VP*H total
  int n = idx >> 9, k = idx & 511;
  float v = (n < VV) ? SW[(size_t)k * VV + n] : 0.0f;
  Wt[(size_t)n * HH + k] = (bf16)v;
}

__global__ void k_bias_pad(const float* __restrict__ sb, float* __restrict__ bpad) {
  int n = blockIdx.x * 256 + threadIdx.x;
  if (n < VP) bpad[n] = (n < VV) ? sb[n] : -1.0e30f;
}

// embedding lookup, time-major: xs[t][b][h] = emb[tokens[b][t]][h], bf16
__global__ void k_embed(const int* __restrict__ tokens, const float* __restrict__ emb,
                        bf16* __restrict__ xs) {
  int idx = blockIdx.x * 256 + threadIdx.x;      // T*B*H total
  int h = idx & 511, b = (idx >> 9) & 15, t = idx >> 13;
  int tok = tokens[b * TT + t];
  xs[idx] = (bf16)emb[(size_t)tok * HH + h];
}

// ---------------- persistent LSTM scan (one workgroup = 32 waves on one WGP) ----------------
// Each wave owns hidden slice u in [wave*16, wave*16+16), computes z tiles for all
// four gates (i,j,f,o) of that slice with 4 WMMA accumulators, then fuses the gates.
// K loop is split into the x-half (K 0..511) and h-half (K 512..1023) so the inner
// loop has no per-iteration source select on the serial critical path. B addresses
// are computed inline from one base so the backend folds the 1 MiB gate strides
// into load immediates.
__device__ inline void lstm_layer(const bf16* __restrict__ xsrc,   // x_t  [16][512] bf16
                                  bf16* __restrict__ hst,          // h state [16][512] (read prev / write new)
                                  const bf16* __restrict__ Wt,     // [2048][1024] bf16
                                  const float* __restrict__ bias,  // [2048]
                                  float* __restrict__ cst,         // c state [16][512] f32
                                  bf16* __restrict__ hs_out,       // optional copy of h_new
                                  int u0, int lane) {
  const int half = lane >> 4, l15 = lane & 15;
  v8f acc[4];
  #pragma unroll
  for (int g = 0; g < 4; ++g) acc[g] = (v8f){0,0,0,0,0,0,0,0};

  const bf16* xrow = xsrc + l15 * HH;
  const bf16* hrow = hst + l15 * HH;
  const bf16* wbase = Wt + (size_t)(u0 + l15) * 1024;   // gate g at +g*512*1024 elems

  // x half: K = 0..511
  for (int kb = 0; kb < 512; kb += 32) {
    v16bf a = load_frag(xrow + kb, half);
    #pragma unroll
    for (int g = 0; g < 4; ++g) {
      v16bf b = load_frag(wbase + (size_t)g * 512 * 1024 + kb, half);
      acc[g] = wmma_bf16(a, b, acc[g]);
    }
  }
  // h half: K = 512..1023
  for (int kb = 0; kb < 512; kb += 32) {
    v16bf a = load_frag(hrow + kb, half);
    #pragma unroll
    for (int g = 0; g < 4; ++g) {
      v16bf b = load_frag(wbase + (size_t)g * 512 * 1024 + 512 + kb, half);
      acc[g] = wmma_bf16(a, b, acc[g]);
    }
  }

  const int u = u0 + l15;
  const float bi = bias[u], bj = bias[512 + u], bf_ = bias[1024 + u], bo = bias[1536 + u];

  __syncthreads();   // all waves done reading previous h before anyone overwrites it

  #pragma unroll
  for (int r = 0; r < 8; ++r) {
    int m = r + 8 * half;                       // D tile: row m, col u
    float zi = acc[0][r] + bi;
    float zj = acc[1][r] + bj;
    float zf = acc[2][r] + bf_;
    float zo = acc[3][r] + bo;
    float co = cst[m * HH + u];
    float cn = co * sigm(zf) + sigm(zi) * tanhf(zj);
    float hn = tanhf(cn) * sigm(zo);
    cst[m * HH + u] = cn;
    bf16 hb = (bf16)hn;
    hst[m * HH + u] = hb;
    if (hs_out) hs_out[m * HH + u] = hb;
  }
  __syncthreads();   // h_new fully written before next layer/timestep reads it
}

__global__ __launch_bounds__(1024) void k_lstm_scan(
    const bf16* __restrict__ xs,  const bf16* __restrict__ W1t, const float* __restrict__ b1,
    const bf16* __restrict__ W2t, const float* __restrict__ b2,
    bf16* __restrict__ hs, bf16* __restrict__ hbuf, float* __restrict__ cbuf) {
  const int tid = threadIdx.x;
  const int wave = tid >> 5, lane = tid & 31;
  const int u0 = wave * 16;
  bf16* h1 = hbuf;
  bf16* h2 = hbuf + BB * HH;
  float* c1 = cbuf;
  float* c2 = cbuf + BB * HH;

  for (int i = tid; i < 2 * BB * HH; i += 1024) { hbuf[i] = (bf16)0.0f; cbuf[i] = 0.0f; }
  __syncthreads();

  for (int t = 0; t < TT; ++t) {
    const bf16* xt = xs + (size_t)t * BB * HH;
    lstm_layer(xt, h1, W1t, b1, c1, nullptr,               u0, lane);  // layer 1
    lstm_layer(h1, h2, W2t, b2, c2, hs + (size_t)t*BB*HH,  u0, lane);  // layer 2 -> hs
  }
}

// ---------------- logits pass 1: per-row max (streamed, no logits materialized) ----------------
// A tile (this wave's 16x512 slice of hs) is hoisted into 16 register fragments and
// reused across the whole n-strip: halves VMEM traffic in the dominant GEMM.
__global__ __launch_bounds__(256) void k_logits_max(
    const bf16* __restrict__ hs, const bf16* __restrict__ Wsmt,
    const float* __restrict__ bpad, unsigned* __restrict__ rowmaxU) {
  int gw = blockIdx.x * 8 + (threadIdx.x >> 5);
  if (gw >= MTILES * NSTRIPS) return;
  const int lane = threadIdx.x & 31, half = lane >> 4, l15 = lane & 15;
  const int mtile = gw / NSTRIPS, strip = gw % NSTRIPS;
  const int n0 = strip * STRIP;
  const int n1 = (n0 + STRIP < NT_V) ? (n0 + STRIP) : NT_V;
  const bf16* arow = hs + (size_t)(mtile * 16 + l15) * HH;

  v16bf afrag[16];
  #pragma unroll
  for (int kk = 0; kk < 16; ++kk) afrag[kk] = load_frag(arow + kk * 32, half);

  float vmax[8];
  #pragma unroll
  for (int r = 0; r < 8; ++r) vmax[r] = -3.0e38f;

  for (int nt = n0; nt < n1; ++nt) {
    const bf16* brow = Wsmt + (size_t)(nt * 16 + l15) * HH;
    v8f acc = (v8f){0,0,0,0,0,0,0,0};
    #pragma unroll
    for (int kk = 0; kk < 16; ++kk) {
      v16bf b = load_frag(brow + kk * 32, half);
      acc = wmma_bf16(afrag[kk], b, acc);
    }
    float vb = bpad[nt * 16 + l15];
    #pragma unroll
    for (int r = 0; r < 8; ++r) vmax[r] = fmaxf(vmax[r], acc[r] + vb);
  }
  #pragma unroll
  for (int r = 0; r < 8; ++r)
    atomicMax(&rowmaxU[mtile * 16 + 8 * half + r], encKey(vmax[r]));
}

__global__ void k_decode_max(const unsigned* __restrict__ u, float* __restrict__ f) {
  int i = blockIdx.x * 256 + threadIdx.x;
  if (i < ROWS) {
    unsigned k = u[i];
    unsigned b = (k & 0x80000000u) ? (k & 0x7FFFFFFFu) : ~k;
    f[i] = __uint_as_float(b);
  }
}

// ---------------- logits pass 2: per-row sum of exp(z - max) ----------------
__global__ __launch_bounds__(256) void k_logits_sumexp(
    const bf16* __restrict__ hs, const bf16* __restrict__ Wsmt,
    const float* __restrict__ bpad, const float* __restrict__ rowmaxF,
    float* __restrict__ rowsum) {
  int gw = blockIdx.x * 8 + (threadIdx.x >> 5);
  if (gw >= MTILES * NSTRIPS) return;
  const int lane = threadIdx.x & 31, half = lane >> 4, l15 = lane & 15;
  const int mtile = gw / NSTRIPS, strip = gw % NSTRIPS;
  const int n0 = strip * STRIP;
  const int n1 = (n0 + STRIP < NT_V) ? (n0 + STRIP) : NT_V;
  const bf16* arow = hs + (size_t)(mtile * 16 + l15) * HH;

  v16bf afrag[16];
  #pragma unroll
  for (int kk = 0; kk < 16; ++kk) afrag[kk] = load_frag(arow + kk * 32, half);

  float rmax[8], vsum[8];
  #pragma unroll
  for (int r = 0; r < 8; ++r) {
    rmax[r] = rowmaxF[mtile * 16 + 8 * half + r];
    vsum[r] = 0.0f;
  }
  for (int nt = n0; nt < n1; ++nt) {
    const bf16* brow = Wsmt + (size_t)(nt * 16 + l15) * HH;
    v8f acc = (v8f){0,0,0,0,0,0,0,0};
    #pragma unroll
    for (int kk = 0; kk < 16; ++kk) {
      v16bf b = load_frag(brow + kk * 32, half);
      acc = wmma_bf16(afrag[kk], b, acc);
    }
    float vb = bpad[nt * 16 + l15];
    #pragma unroll
    for (int r = 0; r < 8; ++r) vsum[r] += __expf(acc[r] + vb - rmax[r]);
  }
  #pragma unroll
  for (int r = 0; r < 8; ++r)
    atomicAdd(&rowsum[mtile * 16 + 8 * half + r], vsum[r]);
}

// ---------------- label logits: one wave per row ----------------
__global__ __launch_bounds__(256) void k_label_logits(
    const bf16* __restrict__ hs, const bf16* __restrict__ Wsmt,
    const float* __restrict__ sb, const int* __restrict__ labels,
    float* __restrict__ ll) {
  int row = blockIdx.x * 8 + (threadIdx.x >> 5);      // row = t*16 + b (hs layout)
  int lane = threadIdx.x & 31;
  int t = row >> 4, b = row & 15;
  int lab = labels[b * TT + t];
  const bf16* h = hs + (size_t)row * HH;
  const bf16* w = Wsmt + (size_t)lab * HH;
  float s = 0.0f;
  for (int k = lane; k < HH; k += 32) s += (float)h[k] * (float)w[k];
  for (int o = 16; o > 0; o >>= 1) s += __shfl_down(s, o, 32);
  if (lane == 0) ll[row] = s + sb[lab];
}

// ---------------- final reduction: cost = sum(max + log(sumexp) - label_logit)/B ----------------
__global__ void k_final_loss(const float* __restrict__ rowmaxF, const float* __restrict__ rowsum,
                             const float* __restrict__ ll, float* __restrict__ out) {
  int i = blockIdx.x * 256 + threadIdx.x;
  if (i < ROWS) {
    float term = (rowmaxF[i] + logf(rowsum[i]) - ll[i]) * (1.0f / (float)BB);
    atomicAdd(out, term);
  }
}

// ---------------- workspace layout (bytes) ----------------
#define OFF_XS     ((size_t)0)                          // T*B*H bf16      = 2 MiB
#define OFF_W1T    (OFF_XS    + (size_t)TT*BB*HH*2)     // 2048*1024 bf16  = 4 MiB
#define OFF_W2T    (OFF_W1T   + (size_t)2048*1024*2)
#define OFF_WSMT   (OFF_W2T   + (size_t)2048*1024*2)    // VP*H bf16       = 51.5 MB
#define OFF_BPAD   (OFF_WSMT  + (size_t)VP*HH*2)        // VP f32
#define OFF_HS     (OFF_BPAD  + (size_t)VP*4)           // T*B*H bf16      = 2 MiB
#define OFF_HBUF   (OFF_HS    + (size_t)TT*BB*HH*2)     // 2*B*H bf16
#define OFF_CBUF   (OFF_HBUF  + (size_t)2*BB*HH*2)      // 2*B*H f32
#define OFF_RMAXU  (OFF_CBUF  + (size_t)2*BB*HH*4)      // 2048 u32
#define OFF_RMAXF  (OFF_RMAXU + (size_t)ROWS*4)
#define OFF_RSUM   (OFF_RMAXF + (size_t)ROWS*4)
#define OFF_LL     (OFF_RSUM  + (size_t)ROWS*4)
// total ~64.4 MB

extern "C" void kernel_launch(void* const* d_in, const int* in_sizes, int n_in,
                              void* d_out, int out_size, void* d_ws, size_t ws_size,
                              hipStream_t stream) {
  (void)in_sizes; (void)n_in; (void)out_size; (void)ws_size;
  const int*   tokens = (const int*)  d_in[0];
  const int*   labels = (const int*)  d_in[1];
  const float* emb    = (const float*)d_in[2];
  const float* W1     = (const float*)d_in[3];
  const float* b1     = (const float*)d_in[4];
  const float* W2     = (const float*)d_in[5];
  const float* b2     = (const float*)d_in[6];
  const float* sw     = (const float*)d_in[7];
  const float* sb     = (const float*)d_in[8];
  float* out = (float*)d_out;

  char* ws = (char*)d_ws;
  bf16*     xs      = (bf16*)    (ws + OFF_XS);
  bf16*     W1t     = (bf16*)    (ws + OFF_W1T);
  bf16*     W2t     = (bf16*)    (ws + OFF_W2T);
  bf16*     Wsmt    = (bf16*)    (ws + OFF_WSMT);
  float*    bpad    = (float*)   (ws + OFF_BPAD);
  bf16*     hs      = (bf16*)    (ws + OFF_HS);
  bf16*     hbuf    = (bf16*)    (ws + OFF_HBUF);
  float*    cbuf    = (float*)   (ws + OFF_CBUF);
  unsigned* rowmaxU = (unsigned*)(ws + OFF_RMAXU);
  float*    rowmaxF = (float*)   (ws + OFF_RMAXF);
  float*    rowsum  = (float*)   (ws + OFF_RSUM);
  float*    ll      = (float*)   (ws + OFF_LL);

  // stats / output init (d_out is poisoned by harness; must zero each call)
  k_init_stats<<<(ROWS + 255) / 256, 256, 0, stream>>>(rowmaxU, rowsum, out);

  // weight conversion / transposition to bf16 N-major
  k_transpose_w <<<(2048 * 1024) / 256, 256, 0, stream>>>(W1, W1t);
  k_transpose_w <<<(2048 * 1024) / 256, 256, 0, stream>>>(W2, W2t);
  k_transpose_sm<<<((size_t)VP * HH) / 256, 256, 0, stream>>>(sw, Wsmt);
  k_bias_pad    <<<(VP + 255) / 256, 256, 0, stream>>>(sb, bpad);

  // embedding lookup (time-major bf16)
  k_embed<<<(TT * BB * HH) / 256, 256, 0, stream>>>(tokens, emb, xs);

  // persistent 2-layer LSTM scan: single workgroup, 32 waves, WMMA + fused gates
  k_lstm_scan<<<1, 1024, 0, stream>>>(xs, W1t, b1, W2t, b2, hs, hbuf, cbuf);

  // streamed log-sum-exp over the [2048 x 50257] logits (never materialized)
  const int passBlocks = (MTILES * NSTRIPS) / 8;   // 1584 blocks * 8 waves
  k_logits_max   <<<passBlocks, 256, 0, stream>>>(hs, Wsmt, bpad, rowmaxU);
  k_decode_max   <<<(ROWS + 255) / 256, 256, 0, stream>>>(rowmaxU, rowmaxF);
  k_logits_sumexp<<<passBlocks, 256, 0, stream>>>(hs, Wsmt, bpad, rowmaxF, rowsum);

  // label logits + final scalar reduction
  k_label_logits<<<ROWS / 8, 256, 0, stream>>>(hs, Wsmt, sb, labels, ll);
  k_final_loss  <<<(ROWS + 255) / 256, 256, 0, stream>>>(rowmaxF, rowsum, ll, out);
}
